// PointNet2PartSegmentationMSG_78151224918842
// MI455X (gfx1250) — matrix operand order repaired and checked by reading, not compiled, not run on hardware
//
#include <hip/hip_runtime.h>

typedef __attribute__((ext_vector_type(16))) _Float16 v16h;
typedef __attribute__((ext_vector_type(8)))  _Float16 v8h;
typedef __attribute__((ext_vector_type(8)))  float    v8f;

#define BATCH 16
#define NPTS  2048

static __host__ __device__ inline int pad32i(int c) { return (c + 31) & ~31; }

// ---------------------------------------------------------------------------
// Farthest point sampling: one block per batch, dist array resident in LDS.
// ---------------------------------------------------------------------------
__global__ void fps_kernel(const float* __restrict__ xyz, int N, int npoint,
                           int* __restrict__ outIdx)
{
    int b = blockIdx.x;
    const float* P = xyz + (size_t)b * N * 3;
    __shared__ float dist[2048];
    __shared__ float sval[256];
    __shared__ int   sidx[256];
    __shared__ int   sfar;
    int tid = threadIdx.x;
    for (int n = tid; n < N; n += blockDim.x) dist[n] = 1e10f;
    if (tid == 0) sfar = 0;
    __syncthreads();
    for (int it = 0; it < npoint; ++it) {
        int far = sfar;
        if (tid == 0) outIdx[b * npoint + it] = far;
        float cx = P[far * 3 + 0], cy = P[far * 3 + 1], cz = P[far * 3 + 2];
        float bestV = -1.0f; int bestI = 0;
        for (int n = tid; n < N; n += blockDim.x) {
            float dx = P[n * 3 + 0] - cx, dy = P[n * 3 + 1] - cy, dz = P[n * 3 + 2] - cz;
            float d = dx * dx + dy * dy + dz * dz;
            float dm = fminf(dist[n], d);
            dist[n] = dm;
            if (dm > bestV) { bestV = dm; bestI = n; }
        }
        sval[tid] = bestV; sidx[tid] = bestI;
        __syncthreads();
        for (int s = blockDim.x / 2; s > 0; s >>= 1) {
            if (tid < s) {
                if (sval[tid + s] > sval[tid] ||
                    (sval[tid + s] == sval[tid] && sidx[tid + s] < sidx[tid])) {
                    sval[tid] = sval[tid + s]; sidx[tid] = sidx[tid + s];
                }
            }
            __syncthreads();
        }
        if (tid == 0) sfar = sidx[0];
        __syncthreads();
    }
}

__global__ void gather_xyz_kernel(const float* __restrict__ xyz, const int* __restrict__ idx,
                                  float* __restrict__ out, int N, int S)
{
    int t = blockIdx.x * blockDim.x + threadIdx.x;
    if (t >= BATCH * S) return;
    int b = t / S;
    int n = idx[t];
    const float* p = xyz + ((size_t)b * N + n) * 3;
    out[(size_t)t * 3 + 0] = p[0];
    out[(size_t)t * 3 + 1] = p[1];
    out[(size_t)t * 3 + 2] = p[2];
}

// ---------------------------------------------------------------------------
// Ball query: ascending index scan == reference's sort-then-take-K semantics.
// ---------------------------------------------------------------------------
__global__ void ball_query_kernel(float r2, int K, const float* __restrict__ xyz,
                                  const float* __restrict__ cent, int* __restrict__ out,
                                  int N, int S)
{
    int t = blockIdx.x * blockDim.x + threadIdx.x;
    if (t >= BATCH * S) return;
    int b = t / S;
    const float* P = xyz + (size_t)b * N * 3;
    float cx = cent[(size_t)t * 3 + 0], cy = cent[(size_t)t * 3 + 1], cz = cent[(size_t)t * 3 + 2];
    int* o = out + (size_t)t * K;
    int cnt = 0, first = 0;
    for (int n = 0; n < N && cnt < K; ++n) {
        float dx = P[n * 3 + 0] - cx, dy = P[n * 3 + 1] - cy, dz = P[n * 3 + 2] - cz;
        float d = dx * dx + dy * dy + dz * dz;
        if (d <= r2) { if (cnt == 0) first = n; o[cnt++] = n; }
    }
    for (int k = cnt; k < K; ++k) o[k] = first;
}

// ---------------------------------------------------------------------------
// Weight prep: fold gamma into W, transpose to cout x Kpad f16, fold bias.
// ---------------------------------------------------------------------------
__global__ void prep_weights_kernel(const float* __restrict__ W, const float* __restrict__ bsrc,
                                    const float* __restrict__ g, const float* __restrict__ be,
                                    _Float16* __restrict__ Wt, float* __restrict__ bias2,
                                    int cin, int cout, int Kpad, int Npad, int hasGB)
{
    int t = blockIdx.x * blockDim.x + threadIdx.x;
    if (t >= Npad * Kpad) return;
    int n = t / Kpad, k = t % Kpad;
    float gg = 1.f, bb = 0.f, bbe = 0.f;
    if (n < cout) {
        if (hasGB) { gg = g[n]; bbe = be[n]; }
        bb = bsrc[n];
    }
    float w = (n < cout && k < cin) ? W[(size_t)k * cout + n] : 0.f;
    Wt[(size_t)n * Kpad + k] = (_Float16)(w * gg);
    if (k == 0) bias2[n] = (n < cout) ? (bb * gg + bbe) : 0.f;
}

// ---------------------------------------------------------------------------
// WMMA GEMM, 2x2 register-blocked: one wave32 computes a 32x32 tile of
// D = relu(A x Wt^T + bias). 8 b128 loads feed 4 back-to-back WMMAs per
// 32-K step (2x arithmetic intensity vs 16x16/wave).
// A: M x Kpad (f16 row-major), Wt: Npad x Kpad (f16, row n = out channel).
// Fragment layout: contiguous K (k = 16*half + s) in both A and B (wave32). M, Npad multiples of 32.
// ---------------------------------------------------------------------------
__global__ void __launch_bounds__(32)
wmma_gemm_kernel(const _Float16* __restrict__ A, const _Float16* __restrict__ Wt,
                 const float* __restrict__ bias, _Float16* __restrict__ Out,
                 float* __restrict__ OutF, int Kpad, int Npad, int relu, int Nreal)
{
    int mt = blockIdx.x, nt = blockIdx.y;
    int lane = threadIdx.x & 31;
    int half = lane >> 4;      // 0: lanes 0-15, 1: lanes 16-31
    int l16  = lane & 15;
    const _Float16* Arow0 = A  + (size_t)(mt * 32 + l16) * Kpad;
    const _Float16* Arow1 = Arow0 + (size_t)16 * Kpad;
    const _Float16* Wrow0 = Wt + (size_t)(nt * 32 + l16) * Kpad;
    const _Float16* Wrow1 = Wrow0 + (size_t)16 * Kpad;
    v8f acc00 = {}, acc01 = {}, acc10 = {}, acc11 = {};
    for (int k = 0; k < Kpad; k += 32) {
        if (k + 64 < Kpad) {
            __builtin_prefetch(Arow0 + k + 64, 0, 3);
            __builtin_prefetch(Arow1 + k + 64, 0, 3);
        }
        // A 16x32 f16: lanes 0-15 hold K [k,k+16); lanes 16-31 K [k+16,k+32)
        v8h a00 = *(const v8h*)(Arow0 + k + 16 * half);
        v8h a01 = *(const v8h*)(Arow0 + k + 16 * half + 8);
        v8h a10 = *(const v8h*)(Arow1 + k + 16 * half);
        v8h a11 = *(const v8h*)(Arow1 + k + 16 * half + 8);
        // B 32x16 f16 (from transposed weights): lanes 0-15 K [k,k+16); 16-31 K [k+16,k+32)
        v8h b00 = *(const v8h*)(Wrow0 + k + 16 * half);
        v8h b01 = *(const v8h*)(Wrow0 + k + 16 * half + 8);
        v8h b10 = *(const v8h*)(Wrow1 + k + 16 * half);
        v8h b11 = *(const v8h*)(Wrow1 + k + 16 * half + 8);
        v16h av0 = __builtin_shufflevector(a00, a01, 0,1,2,3,4,5,6,7,8,9,10,11,12,13,14,15);
        v16h av1 = __builtin_shufflevector(a10, a11, 0,1,2,3,4,5,6,7,8,9,10,11,12,13,14,15);
        v16h bv0 = __builtin_shufflevector(b00, b01, 0,1,2,3,4,5,6,7,8,9,10,11,12,13,14,15);
        v16h bv1 = __builtin_shufflevector(b10, b11, 0,1,2,3,4,5,6,7,8,9,10,11,12,13,14,15);
        acc00 = __builtin_amdgcn_wmma_f32_16x16x32_f16(false, av0, false, bv0, (short)0, acc00, false, false);
        acc01 = __builtin_amdgcn_wmma_f32_16x16x32_f16(false, av0, false, bv1, (short)0, acc01, false, false);
        acc10 = __builtin_amdgcn_wmma_f32_16x16x32_f16(false, av1, false, bv0, (short)0, acc10, false, false);
        acc11 = __builtin_amdgcn_wmma_f32_16x16x32_f16(false, av1, false, bv1, (short)0, acc11, false, false);
    }
    // Epilogue: D layout (per tile): VGPR r -> row r (lanes 0-15) / row 8+r; col = l16.
#pragma unroll
    for (int tm = 0; tm < 2; ++tm) {
#pragma unroll
        for (int tn = 0; tn < 2; ++tn) {
            const v8f& acc = tm == 0 ? (tn == 0 ? acc00 : acc01)
                                     : (tn == 0 ? acc10 : acc11);
            int col = nt * 32 + tn * 16 + l16;
            float bval = bias[col];
#pragma unroll
            for (int r = 0; r < 8; ++r) {
                int row = mt * 32 + tm * 16 + r + 8 * half;
                float v = acc[r] + bval;
                if (relu) v = fmaxf(v, 0.f);
                if (OutF) {
                    if (col < Nreal) OutF[(size_t)row * Nreal + col] = v;
                } else {
                    Out[(size_t)row * Npad + col] = (_Float16)v;
                }
            }
        }
    }
}

// ---------------------------------------------------------------------------
// Group builders (A matrices, zero-padded to Kpad) and max-pool reducers.
// ---------------------------------------------------------------------------
__global__ void build_sa1_kernel(const float* __restrict__ xyz, const float* __restrict__ cxyz,
                                 const int* __restrict__ bq, _Float16* __restrict__ A,
                                 int b, int N, int S, int K, int Kpad)
{
    int r = blockIdx.x * blockDim.x + threadIdx.x;
    if (r >= S * K) return;
    int s = r / K;
    int n = bq[((size_t)b * S + s) * K + (r % K)];
    const float* p = xyz + ((size_t)b * N + n) * 3;
    const float* c = cxyz + ((size_t)b * S + s) * 3;
    _Float16* a = A + (size_t)r * Kpad;
    a[0] = (_Float16)p[0]; a[1] = (_Float16)p[1]; a[2] = (_Float16)p[2];
    a[3] = (_Float16)(p[0] - c[0]); a[4] = (_Float16)(p[1] - c[1]); a[5] = (_Float16)(p[2] - c[2]);
    for (int k = 6; k < Kpad; ++k) a[k] = (_Float16)0.f;
}

__global__ void build_sa2_kernel(const _Float16* __restrict__ feat, int Cf,
                                 const float* __restrict__ xyz, const float* __restrict__ cxyz,
                                 const int* __restrict__ bq, _Float16* __restrict__ A,
                                 int b, int N, int S, int K, int Kpad)
{
    int r = blockIdx.x * blockDim.x + threadIdx.x;
    if (r >= S * K) return;
    int s = r / K;
    int n = bq[((size_t)b * S + s) * K + (r % K)];
    _Float16* a = A + (size_t)r * Kpad;
    const _Float16* f = feat + ((size_t)b * N + n) * Cf;
    for (int c = 0; c < Cf; ++c) a[c] = f[c];
    const float* p  = xyz  + ((size_t)b * N + n) * 3;
    const float* c0 = cxyz + ((size_t)b * S + s) * 3;
    a[Cf + 0] = (_Float16)(p[0] - c0[0]);
    a[Cf + 1] = (_Float16)(p[1] - c0[1]);
    a[Cf + 2] = (_Float16)(p[2] - c0[2]);
    for (int c = Cf + 3; c < Kpad; ++c) a[c] = (_Float16)0.f;
}

__global__ void maxpool_group_kernel(const _Float16* __restrict__ X, int S, int K, int Npad,
                                     int C, _Float16* __restrict__ out, int strideOut,
                                     int chOff, int b)
{
    int t = blockIdx.x * blockDim.x + threadIdx.x;
    if (t >= S * C) return;
    int s = t / C, c = t % C;
    const _Float16* x = X + (size_t)s * K * Npad + c;
    float m = -1e30f;
    for (int k = 0; k < K; ++k) m = fmaxf(m, (float)x[(size_t)k * Npad]);
    out[((size_t)b * S + s) * strideOut + chOff + c] = (_Float16)m;
}

__global__ void build_sa3_kernel(const float* __restrict__ xyz2, const _Float16* __restrict__ f2,
                                 _Float16* __restrict__ A)
{
    int t = blockIdx.x * blockDim.x + threadIdx.x;
    if (t >= BATCH * 128) return;
    _Float16* a = A + (size_t)t * 544;
    const float* p = xyz2 + (size_t)t * 3;
    a[0] = (_Float16)p[0]; a[1] = (_Float16)p[1]; a[2] = (_Float16)p[2];
    const _Float16* f = f2 + (size_t)t * 512;
    for (int c = 0; c < 512; ++c) a[3 + c] = f[c];
    for (int c = 515; c < 544; ++c) a[c] = (_Float16)0.f;
}

__global__ void maxpool_all_kernel(const _Float16* __restrict__ X, _Float16* __restrict__ f3)
{
    int t = blockIdx.x * blockDim.x + threadIdx.x;
    if (t >= BATCH * 1024) return;
    int b = t >> 10, c = t & 1023;
    const _Float16* x = X + (size_t)b * 128 * 1024 + c;
    float m = -1e30f;
    for (int k = 0; k < 128; ++k) m = fmaxf(m, (float)x[(size_t)k * 1024]);
    f3[t] = (_Float16)m;
}

__global__ void build_fp3_kernel(const _Float16* __restrict__ f2, const _Float16* __restrict__ f3,
                                 _Float16* __restrict__ A)
{
    int t = blockIdx.x * blockDim.x + threadIdx.x;
    if (t >= BATCH * 128) return;
    int b = t >> 7;
    _Float16* a = A + (size_t)t * 1536;
    const _Float16* f = f2 + (size_t)t * 512;
    for (int c = 0; c < 512; ++c) a[c] = f[c];
    const _Float16* g = f3 + (size_t)b * 1024;
    for (int c = 0; c < 1024; ++c) a[512 + c] = g[c];
}

__device__ __forceinline__ void knn3(const float* q, const float* P, int S,
                                     int& i0, int& i1, int& i2,
                                     float& w0, float& w1, float& w2)
{
    float d0 = 1e30f, d1 = 1e30f, d2 = 1e30f;
    i0 = i1 = i2 = 0;
    for (int j = 0; j < S; ++j) {
        float dx = P[j * 3 + 0] - q[0], dy = P[j * 3 + 1] - q[1], dz = P[j * 3 + 2] - q[2];
        float d = fmaxf(dx * dx + dy * dy + dz * dz, 0.f);
        if (d < d0)      { d2 = d1; i2 = i1; d1 = d0; i1 = i0; d0 = d; i0 = j; }
        else if (d < d1) { d2 = d1; i2 = i1; d1 = d;  i1 = j; }
        else if (d < d2) { d2 = d;  i2 = j; }
    }
    w0 = 1.f / (d0 + 1e-8f); w1 = 1.f / (d1 + 1e-8f); w2 = 1.f / (d2 + 1e-8f);
    float s = w0 + w1 + w2;
    w0 /= s; w1 /= s; w2 /= s;
}

__global__ void knn_build_fp2_kernel(const float* __restrict__ xyz1, const float* __restrict__ xyz2,
                                     const _Float16* __restrict__ f1, const _Float16* __restrict__ f2b,
                                     _Float16* __restrict__ A)
{
    int t = blockIdx.x * blockDim.x + threadIdx.x;
    if (t >= BATCH * 512) return;
    int b = t >> 9;
    int i0, i1, i2; float w0, w1, w2;
    knn3(xyz1 + (size_t)t * 3, xyz2 + (size_t)b * 128 * 3, 128, i0, i1, i2, w0, w1, w2);
    _Float16* a = A + (size_t)t * 576;
    const _Float16* f = f1 + (size_t)t * 320;
    for (int c = 0; c < 320; ++c) a[c] = f[c];
    const _Float16* p0 = f2b + ((size_t)b * 128 + i0) * 256;
    const _Float16* p1 = f2b + ((size_t)b * 128 + i1) * 256;
    const _Float16* p2 = f2b + ((size_t)b * 128 + i2) * 256;
    for (int c = 0; c < 256; ++c)
        a[320 + c] = (_Float16)(w0 * (float)p0[c] + w1 * (float)p1[c] + w2 * (float)p2[c]);
}

__global__ void knn_build_fp1_kernel(const float* __restrict__ xyz, const float* __restrict__ xyz1,
                                     const float* __restrict__ oh, const _Float16* __restrict__ f1b,
                                     _Float16* __restrict__ A)
{
    int t = blockIdx.x * blockDim.x + threadIdx.x;
    if (t >= BATCH * 2048) return;
    int b = t >> 11;
    int i0, i1, i2; float w0, w1, w2;
    knn3(xyz + (size_t)t * 3, xyz1 + (size_t)b * 512 * 3, 512, i0, i1, i2, w0, w1, w2);
    _Float16* a = A + (size_t)t * 160;
    for (int c = 0; c < 16; ++c) a[c] = (_Float16)oh[b * 16 + c];
    const float* p = xyz + (size_t)t * 3;
    a[16] = (_Float16)p[0]; a[17] = (_Float16)p[1]; a[18] = (_Float16)p[2];
    a[19] = a[16]; a[20] = a[17]; a[21] = a[18];
    const _Float16* q0 = f1b + ((size_t)b * 512 + i0) * 128;
    const _Float16* q1 = f1b + ((size_t)b * 512 + i1) * 128;
    const _Float16* q2 = f1b + ((size_t)b * 512 + i2) * 128;
    for (int c = 0; c < 128; ++c)
        a[22 + c] = (_Float16)(w0 * (float)q0[c] + w1 * (float)q1[c] + w2 * (float)q2[c]);
    for (int c = 150; c < 160; ++c) a[c] = (_Float16)0.f;
}

// ---------------------------------------------------------------------------
// Host orchestration.
// ---------------------------------------------------------------------------
extern "C" void kernel_launch(void* const* d_in, const int* in_sizes, int n_in,
                              void* d_out, int out_size, void* d_ws, size_t ws_size,
                              hipStream_t stream)
{
    (void)in_sizes; (void)n_in; (void)out_size; (void)ws_size;
    const float* x  = (const float*)d_in[0];
    const float* oh = (const float*)d_in[1];

    char* base = (char*)d_ws;
    size_t off = 0;
    auto alloc = [&](size_t bytes) -> void* {
        void* p = base + off;
        off += (bytes + 255) & ~(size_t)255;
        return p;
    };

    struct LDsc { int widx, cin, cout, hasGB; };
    const LDsc LD[26] = {
        {2,6,32,1},{6,32,32,1},{10,32,64,1},          // sa1 b0
        {14,6,64,1},{18,64,64,1},{22,64,128,1},       // sa1 b1
        {26,6,64,1},{30,64,96,1},{34,96,128,1},       // sa1 b2
        {38,323,128,1},{42,128,128,1},{46,128,256,1}, // sa2 b0
        {50,323,128,1},{54,128,196,1},{58,196,256,1}, // sa2 b1
        {62,515,256,1},{66,256,512,1},{70,512,1024,1},// sa3
        {74,1536,256,1},{78,256,256,1},               // fp3
        {82,576,256,1},{86,256,128,1},                // fp2
        {90,150,128,1},{94,128,128,1},                // fp1
        {98,128,128,1},                               // head1
        {102,128,10,0},                               // head2
    };
    _Float16* Wt[26]; float* bias2[26]; int Kp[26], Np[26];
    for (int i = 0; i < 26; ++i) {
        Kp[i] = pad32i(LD[i].cin);
        Np[i] = pad32i(LD[i].cout);
        Wt[i]    = (_Float16*)alloc((size_t)Kp[i] * Np[i] * sizeof(_Float16));
        bias2[i] = (float*)alloc((size_t)Np[i] * sizeof(float));
    }

    int*       idxf1 = (int*)alloc((size_t)BATCH * 512 * 4);
    float*     xyz1  = (float*)alloc((size_t)BATCH * 512 * 3 * 4);
    int*       idxf2 = (int*)alloc((size_t)BATCH * 128 * 4);
    float*     xyz2  = (float*)alloc((size_t)BATCH * 128 * 3 * 4);
    int*       bq    = (int*)alloc((size_t)BATCH * 512 * 128 * 4);
    _Float16*  f1    = (_Float16*)alloc((size_t)BATCH * 512 * 320 * 2);
    _Float16*  f2    = (_Float16*)alloc((size_t)BATCH * 128 * 512 * 2);
    _Float16*  f3    = (_Float16*)alloc((size_t)BATCH * 1024 * 2);
    _Float16*  f2b   = (_Float16*)alloc((size_t)BATCH * 128 * 256 * 2);
    _Float16*  f1b   = (_Float16*)alloc((size_t)BATCH * 512 * 128 * 2);
    _Float16*  ping  = (_Float16*)alloc((size_t)65536 * 128 * 2);
    _Float16*  pong  = (_Float16*)alloc((size_t)65536 * 128 * 2);

    // Weight prep (deterministic each call; graph-capture safe)
    for (int i = 0; i < 26; ++i) {
        const float* W  = (const float*)d_in[LD[i].widx];
        const float* bb = (const float*)d_in[LD[i].widx + 1];
        const float* g  = LD[i].hasGB ? (const float*)d_in[LD[i].widx + 2] : nullptr;
        const float* be = LD[i].hasGB ? (const float*)d_in[LD[i].widx + 3] : nullptr;
        int tot = Np[i] * Kp[i];
        prep_weights_kernel<<<(tot + 255) / 256, 256, 0, stream>>>(
            W, bb, g, be, Wt[i], bias2[i], LD[i].cin, LD[i].cout, Kp[i], Np[i], LD[i].hasGB);
    }

    auto gemm = [&](const _Float16* Ain, int li, _Float16* Oh, float* Of,
                    int M, int relu, int Nreal) {
        dim3 grid(M / 32, Np[li] / 32);   // 32x32 tile per wave (2x2 WMMA block)
        wmma_gemm_kernel<<<grid, 32, 0, stream>>>(Ain, Wt[li], bias2[li], Oh, Of,
                                                  Kp[li], Np[li], relu, Nreal);
    };

    // ---- SA1 (MSG: 2048 -> 512 centers, 3 radii) ----
    fps_kernel<<<BATCH, 256, 0, stream>>>(x, NPTS, 512, idxf1);
    gather_xyz_kernel<<<(BATCH * 512 + 255) / 256, 256, 0, stream>>>(x, idxf1, xyz1, NPTS, 512);

    const float r1[3]  = {0.1f, 0.2f, 0.4f};
    const int   K1[3]  = {32, 64, 128};
    const int   l01[3] = {0, 3, 6};
    const int   co1[3] = {64, 128, 128};
    const int   of1[3] = {0, 64, 192};
    for (int br = 0; br < 3; ++br) {
        int K = K1[br];
        ball_query_kernel<<<(BATCH * 512 + 255) / 256, 256, 0, stream>>>(
            r1[br] * r1[br], K, x, xyz1, bq, NPTS, 512);
        for (int b = 0; b < BATCH; ++b) {
            int rows = 512 * K;
            build_sa1_kernel<<<(rows + 255) / 256, 256, 0, stream>>>(
                x, xyz1, bq, ping, b, NPTS, 512, K, 32);
            gemm(ping, l01[br] + 0, pong, nullptr, rows, 1, 0);
            gemm(pong, l01[br] + 1, ping, nullptr, rows, 1, 0);
            gemm(ping, l01[br] + 2, pong, nullptr, rows, 1, 0);
            int tot = 512 * co1[br];
            maxpool_group_kernel<<<(tot + 255) / 256, 256, 0, stream>>>(
                pong, 512, K, Np[l01[br] + 2], co1[br], f1, 320, of1[br], b);
        }
    }

    // ---- SA2 (MSG: 512 -> 128 centers, 2 radii) ----
    fps_kernel<<<BATCH, 256, 0, stream>>>(xyz1, 512, 128, idxf2);
    gather_xyz_kernel<<<(BATCH * 128 + 255) / 256, 256, 0, stream>>>(xyz1, idxf2, xyz2, 512, 128);

    const float r2a[2] = {0.4f, 0.8f};
    const int   K2[2]  = {64, 128};
    const int   l02[2] = {9, 12};
    const int   of2[2] = {0, 256};
    for (int br = 0; br < 2; ++br) {
        int K = K2[br];
        ball_query_kernel<<<(BATCH * 128 + 255) / 256, 256, 0, stream>>>(
            r2a[br] * r2a[br], K, xyz1, xyz2, bq, 512, 128);
        for (int b = 0; b < BATCH; ++b) {
            int rows = 128 * K;
            build_sa2_kernel<<<(rows + 255) / 256, 256, 0, stream>>>(
                f1, 320, xyz1, xyz2, bq, ping, b, 512, 128, K, 352);
            gemm(ping, l02[br] + 0, pong, nullptr, rows, 1, 0);
            gemm(pong, l02[br] + 1, ping, nullptr, rows, 1, 0);
            gemm(ping, l02[br] + 2, pong, nullptr, rows, 1, 0);
            int tot = 128 * 256;
            maxpool_group_kernel<<<(tot + 255) / 256, 256, 0, stream>>>(
                pong, 128, K, Np[l02[br] + 2], 256, f2, 512, of2[br], b);
        }
    }

    // ---- SA3 (group-all) ----
    build_sa3_kernel<<<(BATCH * 128 + 255) / 256, 256, 0, stream>>>(xyz2, f2, ping);
    gemm(ping, 15, pong, nullptr, BATCH * 128, 1, 0);
    gemm(pong, 16, ping, nullptr, BATCH * 128, 1, 0);
    gemm(ping, 17, pong, nullptr, BATCH * 128, 1, 0);
    maxpool_all_kernel<<<(BATCH * 1024 + 255) / 256, 256, 0, stream>>>(pong, f3);

    // ---- FP3 (broadcast f3, concat f2) ----
    build_fp3_kernel<<<(BATCH * 128 + 255) / 256, 256, 0, stream>>>(f2, f3, ping);
    gemm(ping, 18, pong, nullptr, BATCH * 128, 1, 0);
    gemm(pong, 19, f2b, nullptr, BATCH * 128, 1, 0);

    // ---- FP2 (3-NN interp 128 -> 512, concat f1) ----
    knn_build_fp2_kernel<<<(BATCH * 512 + 255) / 256, 256, 0, stream>>>(xyz1, xyz2, f1, f2b, ping);
    gemm(ping, 20, pong, nullptr, BATCH * 512, 1, 0);
    gemm(pong, 21, f1b, nullptr, BATCH * 512, 1, 0);

    // ---- FP1 (3-NN interp 512 -> 2048, concat [onehot,xyz,xyz]) + heads ----
    knn_build_fp1_kernel<<<(BATCH * 2048 + 255) / 256, 256, 0, stream>>>(x, xyz1, oh, f1b, ping);
    gemm(ping, 22, pong, nullptr, BATCH * 2048, 1, 0);
    gemm(pong, 23, ping, nullptr, BATCH * 2048, 1, 0);
    gemm(ping, 24, pong, nullptr, BATCH * 2048, 1, 0);
    gemm(pong, 25, nullptr, (float*)d_out, BATCH * 2048, 0, 10);  // 128 -> 10, f32 out
}
